// MultiHeadAttention_47974784696872
// MI455X (gfx1250) — matrix operand order repaired and checked by reading, not compiled, not run on hardware
//
#include <hip/hip_runtime.h>

#define DEV static __device__ __forceinline__

typedef __attribute__((ext_vector_type(16))) __bf16 v16bf;
typedef __attribute__((ext_vector_type(8)))  __bf16 v8bf;
typedef __attribute__((ext_vector_type(8)))  float  v8f;
typedef __attribute__((ext_vector_type(4)))  unsigned u32x4;
typedef __attribute__((ext_vector_type(8)))  int      i32x8;
typedef __attribute__((ext_vector_type(4)))  int      i32x4;

#define BATCH   2
#define SEQ     2048
#define DMODEL  1024
#define NHEADS  16
#define DHEAD   64
#define BH      (BATCH*NHEADS)      // 32
#define MROWS   (BATCH*SEQ)         // 4096
#define N3      (3*DMODEL)          // 3072

// GEMM block tiling: 64(M) x 128(N), K-step 32, 8 waves (4 M x 2 N),
// each wave accumulates 16x64 (4 C-fragments, A-fragment reused 4x).
#define BM 64
#define BN 128
#define BK 32
#define A_ELEMS  (BM*BK)            // 2048 bf16 = 4 KB
#define B_ELEMS  (BN*BK)            // 4096 bf16 = 8 KB
#define BUF_ELEMS (A_ELEMS + B_ELEMS)
#define BUF_BYTES (BUF_ELEMS*2)     // 12288

DEV __bf16 f2bf(float f) { return (__bf16)f; }

DEV v16bf make_frag(const __bf16* p0, const __bf16* p1) {
    v8bf lo = *(const v8bf*)p0;
    v8bf hi = *(const v8bf*)p1;
    v16bf r;
#pragma unroll
    for (int i = 0; i < 8; ++i) { r[i] = lo[i]; r[8 + i] = hi[i]; }
    return r;
}

DEV v8f wmma_bf16(v16bf a, v16bf b, v8f c) {
    return __builtin_amdgcn_wmma_f32_16x16x32_bf16(
        false, a, false, b, (short)0, c, false, false);
}

// Async VMEM -> LDS copy (ASYNCcnt path): 16 bytes, GV addressing.
DEV void async_copy_b128(unsigned lds_off, const void* gaddr) {
    asm volatile("global_load_async_to_lds_b128 %0, %1, off"
                 :: "v"(lds_off), "v"(gaddr) : "memory");
}
DEV void wait_async0() {
    asm volatile("s_wait_asynccnt 0x0" ::: "memory");
}

// Tensor Data Mover: DMA a 2-D bf16 tile (BK elems/row x tile_rows, row
// stride DMODEL elems) from global to LDS. TENSORcnt path, one op per tile.
DEV void tdm_load_2d(unsigned lds_off, const void* gptr, int tile_rows) {
    const unsigned long long ga = (unsigned long long)(uintptr_t)gptr;
    u32x4 g0;
    g0[0] = 1u;                                    // count=1 (valid descriptor)
    g0[1] = lds_off;                               // LDS byte address
    g0[2] = (unsigned)ga;                          // global addr [31:0]
    g0[3] = (unsigned)((ga >> 32) & 0x01FFFFFFu)   // global addr [56:32]
          | 0x80000000u;                           // type=2 ("image")
    i32x8 g1;
    g1[0] = 0x00010000;                            // data_size=1 (2 bytes), mask=0
    g1[1] = 0;                                     // tensor_dim0 = 1<<20 (lo16 -> 0)
    g1[2] = 0x0010;                                // tensor_dim0 hi | tensor_dim1 lo
    g1[3] = 0x0010 | (BK << 16);                   // tensor_dim1 hi | tile_dim0=32
    g1[4] = tile_rows;                             // tile_dim1 (tile_dim2=0)
    g1[5] = DMODEL;                                // tensor_dim0_stride (elems)
    g1[6] = 0;
    g1[7] = 0;
    const i32x4 z4 = {};
#if __clang_major__ >= 23
    const i32x8 z8 = {};
    __builtin_amdgcn_tensor_load_to_lds(g0, g1, z4, z4, z8, 0);
#else
    __builtin_amdgcn_tensor_load_to_lds(g0, g1, z4, z4, 0);
#endif
}

// ---------------------------------------------------------------- converters
__global__ void cvt_f32_bf16(const float* __restrict__ src,
                             __bf16* __restrict__ dst, int n) {
    int i = blockIdx.x * blockDim.x + threadIdx.x;
    if (i < n) dst[i] = f2bf(src[i]);
}

// dst[c][r] = src[r][c] (fp32 -> bf16), rows/cols multiples of 32, 256 threads
__global__ void transpose_to_bf16(const float* __restrict__ src,
                                  __bf16* __restrict__ dst,
                                  int rows, int cols) {
    __shared__ float tile[32][33];
    const int c0 = blockIdx.x * 32, r0 = blockIdx.y * 32;
    const int tx = threadIdx.x & 31, ty = threadIdx.x >> 5;
#pragma unroll
    for (int i = ty; i < 32; i += 8)
        tile[i][tx] = src[(size_t)(r0 + i) * cols + (c0 + tx)];
    __syncthreads();
#pragma unroll
    for (int i = ty; i < 32; i += 8)
        dst[(size_t)(c0 + i) * rows + (r0 + tx)] = f2bf(tile[tx][i]);
}

// ---------------------------------------------------------------- GEMM core
// acc[4] = 16x64 per wave for C tile (m0..m0+63, n0..n0+127).
// A [M,K] bf16 row-major; Bt [N,K] bf16 row-major. K = DMODEL.
// Double-buffered LDS staging: USE_TDM ? tensor_load_to_lds (TENSORcnt)
//                                      : global_load_async_to_lds (ASYNCcnt)
template <bool USE_TDM>
DEV void gemm_tile(const __bf16* __restrict__ A_,
                   const __bf16* __restrict__ Bt_,
                   int m0, int n0, __bf16* smem, v8f acc[4]) {
    const int t    = threadIdx.x;
    const int lane = t & 31;
    const int wave = t >> 5;
    const int ln = lane & 15, half = lane >> 4;
    const int wm = wave >> 1, wn = wave & 1;   // 4 x 2 wave grid
    const unsigned lbase = (unsigned)(uintptr_t)smem;

    // per-thread async staging: A = 256 x b128 (1/thread), B = 512 (2/thread)
    const int rA = t >> 2, cA = t & 3;
    const __bf16* gA  = A_  + (size_t)(m0 + rA) * DMODEL + cA * 8;
    const __bf16* gB0 = Bt_ + (size_t)(n0 + rA) * DMODEL + cA * 8;
    const __bf16* gB1 = gB0 + (size_t)64 * DMODEL;
    const unsigned lA  = lbase + (unsigned)(rA * BK + cA * 8) * 2;
    const unsigned lB0 = lbase + (unsigned)(A_ELEMS + rA * BK + cA * 8) * 2;
    const unsigned lB1 = lB0 + 64 * BK * 2;

    // compute-side LDS pointers
    const __bf16* sA0 = smem + (wm * 16 + ln) * BK + half * 16;
    const __bf16* sB0 = smem + A_ELEMS + (wn * 64 + ln) * BK + half * 16;

    // prologue: stage k-step 0 into buffer 0
    if (USE_TDM) {
        if (wave == 0) {
            tdm_load_2d(lbase, A_ + (size_t)m0 * DMODEL, BM);
            tdm_load_2d(lbase + A_ELEMS * 2, Bt_ + (size_t)n0 * DMODEL, BN);
        }
    } else {
        async_copy_b128(lA, gA);
        async_copy_b128(lB0, gB0);
        async_copy_b128(lB1, gB1);
    }

    const int NK = DMODEL / BK;
    for (int kt = 0; kt < NK; ++kt) {
        if (USE_TDM) {
            if (wave == 0) {
                if (kt + 1 < NK) {   // issue next buffer, wait for current (2 in flight)
                    const unsigned bo = (unsigned)(((kt + 1) & 1) * BUF_BYTES);
                    const int ko = (kt + 1) * BK;
                    tdm_load_2d(lbase + bo, A_ + (size_t)m0 * DMODEL + ko, BM);
                    tdm_load_2d(lbase + bo + A_ELEMS * 2,
                                Bt_ + (size_t)n0 * DMODEL + ko, BN);
                    __builtin_amdgcn_s_wait_tensorcnt(2);
                } else {
                    __builtin_amdgcn_s_wait_tensorcnt(0);
                }
            }
            __syncthreads();
        } else {
            wait_async0();
            __syncthreads();
            if (kt + 1 < NK) {
                const unsigned bo = (unsigned)(((kt + 1) & 1) * BUF_BYTES);
                const int ko = (kt + 1) * BK;
                async_copy_b128(lA + bo, gA + ko);
                async_copy_b128(lB0 + bo, gB0 + ko);
                async_copy_b128(lB1 + bo, gB1 + ko);
            }
        }
        const int be = (kt & 1) * BUF_ELEMS;
        const __bf16* pa = sA0 + be;
        const v16bf af = make_frag(pa, pa + 8);
#pragma unroll
        for (int f = 0; f < 4; ++f) {
            const __bf16* pb = sB0 + be + f * 16 * BK;
            acc[f] = wmma_bf16(af, make_frag(pb, pb + 8), acc[f]);
        }
        __syncthreads();
    }
}

// ---------------------------------------------------------------- QKV GEMM
// ASYNCcnt-staged. Epilogue: +bias, route column n -> (h=n/192, dh=(n%192)/3,
// c=n%3) into Q (scaled 1/8) [bh,s,dh], K [bh,s,dh], V transposed [bh,dh,s].
__global__ void __launch_bounds__(256)
qkv_gemm(const __bf16* __restrict__ A,
         const __bf16* __restrict__ Bt,
         const float*  __restrict__ bias,
         __bf16* __restrict__ Qb,
         __bf16* __restrict__ Kb,
         __bf16* __restrict__ Vt) {
    __shared__ __bf16 smem[2 * BUF_ELEMS];
    const int lane = threadIdx.x & 31, wave = threadIdx.x >> 5;
    const int ln = lane & 15, half = lane >> 4;
    const int wm = wave >> 1, wn = wave & 1;
    const int m0 = blockIdx.y * BM, n0 = blockIdx.x * BN;
    v8f acc[4] = {{}, {}, {}, {}};
    gemm_tile<false>(A, Bt, m0, n0, smem, acc);
#pragma unroll
    for (int f = 0; f < 4; ++f) {
        const int n  = n0 + wn * 64 + f * 16 + ln;
        const float bv = bias[n];
        const int h  = n / 192;
        const int rm = n - h * 192;
        const int dh = rm / 3;
        const int c  = rm - dh * 3;
#pragma unroll
        for (int j = 0; j < 8; ++j) {
            const int m  = m0 + wm * 16 + j + half * 8;
            const int b  = m >> 11;           // / SEQ
            const int s  = m & (SEQ - 1);
            const int bh = b * NHEADS + h;
            const float v = acc[f][j] + bv;
            if (c == 0)      Qb[((size_t)bh * SEQ + s) * DHEAD + dh] = f2bf(v * 0.125f);
            else if (c == 1) Kb[((size_t)bh * SEQ + s) * DHEAD + dh] = f2bf(v);
            else             Vt[((size_t)bh * DHEAD + dh) * SEQ + s] = f2bf(v);
        }
    }
}

// ---------------------------------------------------------------- output GEMM
// TDM-staged (TENSORcnt).
__global__ void __launch_bounds__(256)
out_gemm(const __bf16* __restrict__ A,     // ctx [MROWS, DMODEL]
         const __bf16* __restrict__ Bt,    // W_out^T [DMODEL, DMODEL]
         const float*  __restrict__ bias,
         float* __restrict__ out) {
    __shared__ __bf16 smem[2 * BUF_ELEMS];
    const int lane = threadIdx.x & 31, wave = threadIdx.x >> 5;
    const int ln = lane & 15, half = lane >> 4;
    const int wm = wave >> 1, wn = wave & 1;
    const int m0 = blockIdx.y * BM, n0 = blockIdx.x * BN;
    v8f acc[4] = {{}, {}, {}, {}};
    gemm_tile<true>(A, Bt, m0, n0, smem, acc);
#pragma unroll
    for (int f = 0; f < 4; ++f) {
        const int n = n0 + wn * 64 + f * 16 + ln;
        const float bv = bias[n];
#pragma unroll
        for (int j = 0; j < 8; ++j) {
            const int m = m0 + wm * 16 + j + half * 8;
            out[(size_t)m * DMODEL + n] = acc[f][j] + bv;
        }
    }
}

// ---------------------------------------------------------------- attention
// One 16-query tile per wave, streaming all keys in 32-wide chunks with
// online softmax (flash style). 8 waves / block.
__global__ void __launch_bounds__(256)
attn(const __bf16* __restrict__ Qb,
     const __bf16* __restrict__ Kb,
     const __bf16* __restrict__ Vt,
     const unsigned char* __restrict__ mask,
     __bf16* __restrict__ ctx) {           // [MROWS, DMODEL]
    __shared__ __bf16 pstage[8 * 16 * 32];
    const int lane = threadIdx.x & 31;
    const int wave = threadIdx.x >> 5;
    const int ln = lane & 15, half = lane >> 4;
    const int tile = blockIdx.x * 8 + wave;
    const int bh = tile >> 7;                 // / (SEQ/16)
    const int q0 = (tile & 127) * 16;

    const __bf16* qrow = Qb + ((size_t)bh * SEQ + q0 + ln) * DHEAD + half * 16;
    const v16bf aq0 = make_frag(qrow,      qrow + 8);
    const v16bf aq1 = make_frag(qrow + 32, qrow + 40);

    v8f o0 = {}, o1 = {}, o2 = {}, o3 = {};
    float mrow[8], lrow[8];
#pragma unroll
    for (int j = 0; j < 8; ++j) { mrow[j] = -1e30f; lrow[j] = 0.f; }

    const unsigned char* mrowp = mask + (size_t)bh * SEQ;
    __bf16* ps = pstage + wave * (16 * 32);

    for (int kv = 0; kv < SEQ; kv += 32) {
        const __bf16* kb0 = Kb + ((size_t)bh * SEQ + kv + ln) * DHEAD + half * 16;
        const __bf16* kb1 = kb0 + 16 * DHEAD;
        v8f s0 = {}, s1 = {};
        s0 = wmma_bf16(aq0, make_frag(kb0,      kb0 + 8),  s0);
        s0 = wmma_bf16(aq1, make_frag(kb0 + 32, kb0 + 40), s0);
        s1 = wmma_bf16(aq0, make_frag(kb1,      kb1 + 8),  s1);
        s1 = wmma_bf16(aq1, make_frag(kb1 + 32, kb1 + 40), s1);

        const bool msk0 = mrowp[kv + ln] != 0;
        const bool msk1 = mrowp[kv + 16 + ln] != 0;
#pragma unroll
        for (int j = 0; j < 8; ++j) {
            if (msk0) s0[j] = -1e30f;
            if (msk1) s1[j] = -1e30f;
        }

#pragma unroll
        for (int j = 0; j < 8; ++j) {
            float mx = fmaxf(s0[j], s1[j]);
            mx = fmaxf(mx, __shfl_xor(mx, 1));
            mx = fmaxf(mx, __shfl_xor(mx, 2));
            mx = fmaxf(mx, __shfl_xor(mx, 4));
            mx = fmaxf(mx, __shfl_xor(mx, 8));
            const float mnew = fmaxf(mrow[j], mx);
            const float sc = __expf(mrow[j] - mnew);
            mrow[j] = mnew;
            const float p0 = __expf(s0[j] - mnew);
            const float p1 = __expf(s1[j] - mnew);
            float rs = p0 + p1;
            rs += __shfl_xor(rs, 1);
            rs += __shfl_xor(rs, 2);
            rs += __shfl_xor(rs, 4);
            rs += __shfl_xor(rs, 8);
            lrow[j] = lrow[j] * sc + rs;
            o0[j] *= sc; o1[j] *= sc; o2[j] *= sc; o3[j] *= sc;
            s0[j] = p0;  s1[j] = p1;
        }

        // stage P (16x32 bf16) through LDS: C-layout (lane=col) -> A-layout (lane=row)
#pragma unroll
        for (int j = 0; j < 8; ++j) {
            const int r = j + half * 8;
            ps[r * 32 + ln]      = f2bf(s0[j]);
            ps[r * 32 + 16 + ln] = f2bf(s1[j]);
        }
        __builtin_amdgcn_wave_barrier();
        const __bf16* prow = ps + ln * 32 + half * 16;
        const v16bf ap = make_frag(prow, prow + 8);
        __builtin_amdgcn_wave_barrier();

        const __bf16* vb = Vt + ((size_t)bh * DHEAD + ln) * SEQ + kv + half * 16;
        o0 = wmma_bf16(ap, make_frag(vb,            vb + 8),            o0);
        o1 = wmma_bf16(ap, make_frag(vb + 16 * SEQ, vb + 16 * SEQ + 8), o1);
        o2 = wmma_bf16(ap, make_frag(vb + 32 * SEQ, vb + 32 * SEQ + 8), o2);
        o3 = wmma_bf16(ap, make_frag(vb + 48 * SEQ, vb + 48 * SEQ + 8), o3);
    }

    const int b = bh / NHEADS, h = bh % NHEADS;
#pragma unroll
    for (int j = 0; j < 8; ++j) {
        const float inv = 1.0f / lrow[j];
        const int srow = q0 + j + half * 8;
        const size_t base = ((size_t)b * SEQ + srow) * DMODEL + h * DHEAD + ln;
        ctx[base +  0] = f2bf(o0[j] * inv);
        ctx[base + 16] = f2bf(o1[j] * inv);
        ctx[base + 32] = f2bf(o2[j] * inv);
        ctx[base + 48] = f2bf(o3[j] * inv);
    }
}

// ---------------------------------------------------------------- launcher
extern "C" void kernel_launch(void* const* d_in, const int* in_sizes, int n_in,
                              void* d_out, int out_size, void* d_ws, size_t ws_size,
                              hipStream_t stream) {
    (void)in_sizes; (void)n_in; (void)out_size; (void)ws_size;
    const float*         query = (const float*)d_in[0];
    const unsigned char* mask  = (const unsigned char*)d_in[1];
    const float*         W_qkv = (const float*)d_in[2];
    const float*         b_qkv = (const float*)d_in[3];
    const float*         W_out = (const float*)d_in[4];
    const float*         b_out = (const float*)d_in[5];
    float* out = (float*)d_out;

    char* ws = (char*)d_ws;
    size_t off = 0;
    __bf16* q16 = (__bf16*)(ws + off); off += (size_t)MROWS * DMODEL * 2;   //  8 MB
    __bf16* w1t = (__bf16*)(ws + off); off += (size_t)N3 * DMODEL * 2;      //  6 MB
    __bf16* w2t = (__bf16*)(ws + off); off += (size_t)DMODEL * DMODEL * 2;  //  2 MB
    __bf16* Qb  = (__bf16*)(ws + off); off += (size_t)BH * SEQ * DHEAD * 2; //  8 MB
    __bf16* Kb  = (__bf16*)(ws + off); off += (size_t)BH * SEQ * DHEAD * 2; //  8 MB
    __bf16* Vt  = (__bf16*)(ws + off); off += (size_t)BH * SEQ * DHEAD * 2; //  8 MB
    __bf16* ctx = (__bf16*)(ws + off); off += (size_t)MROWS * DMODEL * 2;   //  8 MB

    cvt_f32_bf16<<<(MROWS * DMODEL) / 256, 256, 0, stream>>>(query, q16, MROWS * DMODEL);
    transpose_to_bf16<<<dim3(N3 / 32, DMODEL / 32), 256, 0, stream>>>(W_qkv, w1t, DMODEL, N3);
    transpose_to_bf16<<<dim3(DMODEL / 32, DMODEL / 32), 256, 0, stream>>>(W_out, w2t, DMODEL, DMODEL);
    qkv_gemm<<<dim3(N3 / BN, MROWS / BM), 256, 0, stream>>>(q16, w1t, b_qkv, Qb, Kb, Vt);
    attn<<<dim3(BH * (SEQ / 16) / 8), 256, 0, stream>>>(Qb, Kb, Vt, mask, ctx);
    out_gemm<<<dim3(DMODEL / BN, MROWS / BM), 256, 0, stream>>>(ctx, w2t, b_out, out);
}
